// AttentionGNN_71768903516531
// MI455X (gfx1250) — matrix operand order repaired and checked by reading, not compiled, not run on hardware
//
#include <hip/hip_runtime.h>
#include <hip/hip_bf16.h>

// ---------------------------------------------------------------------------
// AttentionGNN forward for MI455X (gfx1250, wave32, WMMA).
//  - Matrix math: v_wmma_f32_16x16x32_f16, fp32 accumulate.
//  - Weights pre-packed once per launch into fragment-ordered fp16 with K
//    zero-padded to 32 -> B fragment = 2 coalesced 16B loads, no LDS; zero
//    pad kills A K-tail garbage (clamped branchless loads).
//  - A fragments loaded straight from global as 4x float4 per lane.
//  - One wave computes an (MB*16) x N block: NTILES x MB accumulators in
//    VGPRs; each B fragment is reused for MB WMMAs. K-tail hoisted out of
//    the main loop (no per-iteration branch).
//  - Edge gather/scatter: wave-per-edge VALU + global_atomic_add_f32; segment
//    max via monotonic-uint atomicMax.
// ---------------------------------------------------------------------------

typedef __attribute__((ext_vector_type(16))) _Float16 v16h;
typedef __attribute__((ext_vector_type(8)))  _Float16 v8h;
typedef __attribute__((ext_vector_type(8)))  float    v8f;

#define NNODE 50000
#define NEDGE 300000
#define LDIM  112     // LABEL_DIM
#define LK    4       // LABEL_K

// ---------------- helpers ----------------
__device__ __forceinline__ float wave_sum(float v) {
#pragma unroll
  for (int off = 16; off; off >>= 1) v += __shfl_xor(v, off, 32);
  return v;
}

__device__ __forceinline__ unsigned fenc(float f) {
  unsigned u = __float_as_uint(f);
  return (u & 0x80000000u) ? ~u : (u | 0x80000000u);
}
__device__ __forceinline__ float fdec(unsigned u) {
  return (u & 0x80000000u) ? __uint_as_float(u & 0x7FFFFFFFu)
                           : __uint_as_float(~u);
}

// ---------------------------------------------------------------------------
// Pack fp32 weight B [K,N] (or [N,K] if transB) into fragment-ordered fp16:
//   P[((tn*nkb + kb)*32 + lane)*16 + j] = B(k = kb*32 + (lane<16?0:8) + (j<8?j:j+8),
//                                          n = tn*16 + (lane&15)),  0 if k>=K
// Matches the CDNA5 16-bit B(32x16) VGPR layout: lane reads its 16 halves as
// one contiguous 32B chunk, consecutive lanes adjacent (perfectly coalesced).
// ---------------------------------------------------------------------------
__global__ void pack_b_kernel(const float* __restrict__ B, int ldb, int transB,
                              _Float16* __restrict__ P, int K, int N)
{
  const int nkb = (K + 31) >> 5;
  const int total = nkb * (N >> 4) * 512;
  int i = blockIdx.x * blockDim.x + threadIdx.x;
  if (i >= total) return;
  int j    = i & 15;
  int lane = (i >> 4) & 31;
  int rest = i >> 9;
  int kb   = rest % nkb;
  int tn   = rest / nkb;
  int col  = tn * 16 + (lane & 15);
  int k    = kb * 32 + ((lane < 16) ? 0 : 8) + ((j < 8) ? j : j + 8);
  float v = 0.f;
  if (k < K)
    v = transB ? B[(size_t)col * ldb + k] : B[(size_t)k * ldb + col];
  P[i] = (_Float16)v;
}

// ---------------------------------------------------------------------------
// Fragment loaders
// ---------------------------------------------------------------------------
__device__ __forceinline__ v16h load_a_frag_full(const float* ap, int k0, int kba)
{
  // two 32B runs matching ISA A(16x32) layout: K = kba..+7 and kba+16..+23
  const float4 a0 = *(const float4*)(ap + k0 + kba);
  const float4 a1 = *(const float4*)(ap + k0 + kba + 4);
  const float4 a2 = *(const float4*)(ap + k0 + kba + 16);
  const float4 a3 = *(const float4*)(ap + k0 + kba + 20);
  v16h af;
  af[0]  = (_Float16)a0.x; af[1]  = (_Float16)a0.y;
  af[2]  = (_Float16)a0.z; af[3]  = (_Float16)a0.w;
  af[4]  = (_Float16)a1.x; af[5]  = (_Float16)a1.y;
  af[6]  = (_Float16)a1.z; af[7]  = (_Float16)a1.w;
  af[8]  = (_Float16)a2.x; af[9]  = (_Float16)a2.y;
  af[10] = (_Float16)a2.z; af[11] = (_Float16)a2.w;
  af[12] = (_Float16)a3.x; af[13] = (_Float16)a3.y;
  af[14] = (_Float16)a3.z; af[15] = (_Float16)a3.w;
  return af;
}

__device__ __forceinline__ v16h load_a_frag_tail(const float* ap, int k0, int kba, int K)
{
  // clamped branchless loads; garbage killed by zero-padded packed B
  v16h af;
#pragma unroll
  for (int j = 0; j < 8; ++j) {
    int k1 = k0 + kba + j;
    int k2 = k0 + kba + 16 + j;
    af[j]     = (_Float16)ap[k1 < K ? k1 : 0];
    af[8 + j] = (_Float16)ap[k2 < K ? k2 : 0];
  }
  return af;
}

__device__ __forceinline__ v16h load_b_frag(const _Float16* pb)
{
  v8h blo = *(const v8h*)(pb);
  v8h bhi = *(const v8h*)(pb + 8);
  return __builtin_shufflevector(blo, bhi,
           0,1,2,3,4,5,6,7,8,9,10,11,12,13,14,15);
}

// ---------------------------------------------------------------------------
// WMMA GEMM: C[M, NTILES*16] = A[M,K] (row-gathered if GATHER) * Bpacked
//            (+bias) (+relu).  One wave per MB 16-row tiles; blockDim = 128.
// Requires: M % 16 == 0, lda % 4 == 0, A base 16B aligned.
// ---------------------------------------------------------------------------
template<int NTILES, int MB, bool GATHER>
__global__ __launch_bounds__(128)
void wmma_gemm_kernel(const float* __restrict__ A, int lda,
                      const int* __restrict__ gidx,
                      const _Float16* __restrict__ Bp,
                      const float* __restrict__ bias,
                      float* __restrict__ Cmat, int ldc,
                      int Mtiles, int K, int relu)
{
  const int lane  = threadIdx.x & 31;
  const int group = blockIdx.x * 4 + (threadIdx.x >> 5);
  const int tm0   = group * MB;
  if (tm0 >= Mtiles) return;

  const int r     = lane & 15;
  const int kba   = (lane < 16) ? 0 : 8;   // A-fragment K base (ISA layout)
  const int nkb   = (K + 31) >> 5;
  const int kfull = K >> 5;

  const float* ap[MB];
  bool valid[MB];
#pragma unroll
  for (int m = 0; m < MB; ++m) {
    int tm   = tm0 + m;
    valid[m] = (tm < Mtiles);
    int gr   = valid[m] ? tm * 16 + r : 0;
    int ar   = GATHER ? gidx[gr] : gr;
    ap[m]    = A + (size_t)ar * lda;
  }

  v8f acc[MB][NTILES];
#pragma unroll
  for (int m = 0; m < MB; ++m)
#pragma unroll
    for (int t = 0; t < NTILES; ++t) acc[m][t] = (v8f){};

  // full 32-K blocks: branch-free inner loop
  for (int kb = 0; kb < kfull; ++kb) {
    const int k0 = kb * 32;
    v16h af[MB];
#pragma unroll
    for (int m = 0; m < MB; ++m) af[m] = load_a_frag_full(ap[m], k0, kba);
#pragma unroll
    for (int t = 0; t < NTILES; ++t) {
      v16h bf = load_b_frag(Bp + (((size_t)t * nkb + kb) * 32 + lane) * 16);
#pragma unroll
      for (int m = 0; m < MB; ++m)
        acc[m][t] = __builtin_amdgcn_wmma_f32_16x16x32_f16(
            false, af[m], false, bf, (short)0, acc[m][t], false, false);
    }
  }
  // K tail (once, outside the hot loop)
  if (K & 31) {
    const int k0 = kfull * 32;
    v16h af[MB];
#pragma unroll
    for (int m = 0; m < MB; ++m) af[m] = load_a_frag_tail(ap[m], k0, kba, K);
#pragma unroll
    for (int t = 0; t < NTILES; ++t) {
      v16h bf = load_b_frag(Bp + (((size_t)t * nkb + kfull) * 32 + lane) * 16);
#pragma unroll
      for (int m = 0; m < MB; ++m)
        acc[m][t] = __builtin_amdgcn_wmma_f32_16x16x32_f16(
            false, af[m], false, bf, (short)0, acc[m][t], false, false);
    }
  }

  // writeback: C/D layout — VGPR rr8: lanes 0-15 -> M=rr8, lanes 16-31 -> rr8+8
  const int c = lane & 15;
#pragma unroll
  for (int m = 0; m < MB; ++m) {
    if (!valid[m]) continue;
    const int row0 = (tm0 + m) * 16;
#pragma unroll
    for (int t = 0; t < NTILES; ++t) {
      int gc = t * 16 + c;
      float b = bias ? bias[gc] : 0.f;
#pragma unroll
      for (int rr8 = 0; rr8 < 8; ++rr8) {
        int rr = (lane < 16) ? rr8 : rr8 + 8;
        float v = acc[m][t][rr8] + b;
        if (relu) v = fmaxf(v, 0.f);
        Cmat[(size_t)(row0 + rr) * ldc + gc] = v;
      }
    }
  }
}

// ---------------------------------------------------------------------------
// Edge kernel 1: feature score + label attention (K=4); wave per edge.
// ---------------------------------------------------------------------------
__global__ __launch_bounds__(256)
void edge_scores_kernel(const int* __restrict__ src, const int* __restrict__ dst,
                        const float* __restrict__ q, const float* __restrict__ kf,
                        const float* __restrict__ ek, const float* __restrict__ y,
                        const float* __restrict__ t,
                        const float* __restrict__ Wltk, const float* __restrict__ bltk,
                        const int* __restrict__ mask,
                        float* __restrict__ sfeat, float* __restrict__ xl,
                        unsigned* __restrict__ nmax,
                        int E, int C)
{
  const int lane = threadIdx.x & 31;
  const int e = blockIdx.x * 8 + (threadIdx.x >> 5);
  if (e >= E) return;
  const int sn = src[e], dn = dst[e];

  float dall = 0.f, dqe = 0.f, qs = 0.f;
  for (int c = lane; c < C; c += 32) {
    float qi = q [(size_t)dn * C + c];
    float kj = kf[(size_t)sn * C + c];
    float ee = ek[(size_t)e  * C + c];
    dall += qi * (ee + kj);
    dqe  += qi * ee;
    qs   += qi;
  }
  float lab0 = 0.f, lab1 = 0.f, lab2 = 0.f, lab3 = 0.f;
  for (int d = lane; d < LDIM; d += 32) {
    float u = y[(size_t)sn * LDIM + d] * t[(size_t)e * LDIM + d];
    lab0 += u * Wltk[d * LK + 0];
    lab1 += u * Wltk[d * LK + 1];
    lab2 += u * Wltk[d * LK + 2];
    lab3 += u * Wltk[d * LK + 3];
  }
  dall = wave_sum(dall); dqe = wave_sum(dqe); qs = wave_sum(qs);
  lab0 = wave_sum(lab0); lab1 = wave_sum(lab1);
  lab2 = wave_sum(lab2); lab3 = wave_sum(lab3);

  if (lane == 0) {
    float sf = dall * rsqrtf((float)C);
    sfeat[e] = sf;
    atomicMax(&nmax[dn], fenc(sf));

    float madd = (mask[sn] > 0) ? -1e9f : 0.f;
    float s0 = 0.5f * (dqe + lab0 + bltk[0] * qs) + madd;
    float s1 = 0.5f * (dqe + lab1 + bltk[1] * qs) + madd;
    float s2 = 0.5f * (dqe + lab2 + bltk[2] * qs) + madd;
    float s3 = 0.5f * (dqe + lab3 + bltk[3] * qs) + madd;
    float mx = fmaxf(fmaxf(s0, s1), fmaxf(s2, s3));
    float e0 = expf(s0 - mx), e1 = expf(s1 - mx);
    float e2 = expf(s2 - mx), e3 = expf(s3 - mx);
    float inv = 1.f / (e0 + e1 + e2 + e3);
    xl[(size_t)e * 4 + 0] = s0 * e0 * inv;
    xl[(size_t)e * 4 + 1] = s1 * e1 * inv;
    xl[(size_t)e * 4 + 2] = s2 * e2 * inv;
    xl[(size_t)e * 4 + 3] = s3 * e3 * inv;
  }
}

// ---------------------------------------------------------------------------
// Edge kernel 2: e = exp(s - max[dst]); atomic denominator per dst.
// ---------------------------------------------------------------------------
__global__ __launch_bounds__(256)
void edge_expsum_kernel(const int* __restrict__ dst,
                        const float* __restrict__ sfeat,
                        const unsigned* __restrict__ nmax,
                        float* __restrict__ eexp, float* __restrict__ den, int E)
{
  int e = blockIdx.x * blockDim.x + threadIdx.x;
  if (e >= E) return;
  int dn = dst[e];
  float ee = expf(sfeat[e] - fdec(nmax[dn]));
  eexp[e] = ee;
  atomicAdd(&den[dn], ee);
}

// ---------------------------------------------------------------------------
// Edge kernel 3: alpha = e/den[dst]; scatter f = v[src]*alpha and
// lmsg = xl @ Wkto + bkto into h2[:, C:3C] with float atomics.
// ---------------------------------------------------------------------------
__global__ __launch_bounds__(256)
void edge_scatter_kernel(const int* __restrict__ src, const int* __restrict__ dst,
                         const float* __restrict__ vf,
                         const float* __restrict__ eexp, const float* __restrict__ den,
                         const float* __restrict__ xl,
                         const float* __restrict__ Wkto, const float* __restrict__ bkto,
                         float* __restrict__ h2, int ldc, int C, int E)
{
  const int lane = threadIdx.x & 31;
  const int e = blockIdx.x * 8 + (threadIdx.x >> 5);
  if (e >= E) return;
  const int sn = src[e], dn = dst[e];
  const float alpha = eexp[e] / den[dn];
  const float x0 = xl[(size_t)e * 4 + 0], x1 = xl[(size_t)e * 4 + 1];
  const float x2 = xl[(size_t)e * 4 + 2], x3 = xl[(size_t)e * 4 + 3];

  for (int c = lane; c < C; c += 32) {
    float f = vf[(size_t)sn * C + c] * alpha;
    atomicAdd(&h2[(size_t)dn * ldc + C + c], f);
    float lm = bkto[c] + x0 * Wkto[c] + x1 * Wkto[C + c]
             + x2 * Wkto[2 * C + c] + x3 * Wkto[3 * C + c];
    atomicAdd(&h2[(size_t)dn * ldc + 2 * C + c], lm);
  }
}

__global__ void fill_u32_kernel(unsigned* __restrict__ p, unsigned v, int n) {
  int i = blockIdx.x * blockDim.x + threadIdx.x;
  if (i < n) p[i] = v;
}

// ---------------------------------------------------------------------------
// Host-side helpers
// ---------------------------------------------------------------------------
static _Float16* pack_weight(const float* B, int ldb, int transB, int K, int N,
                             _Float16*& arena, hipStream_t s)
{
  const int nkb = (K + 31) / 32, ntn = N / 16;
  const int total = nkb * ntn * 512;
  _Float16* P = arena;
  arena += total;
  pack_b_kernel<<<(total + 255) / 256, 256, 0, s>>>(B, ldb, transB, P, K, N);
  return P;
}

static void launch_gemm(const float* A, int lda, const int* gidx,
                        const _Float16* Bp, const float* bias,
                        float* Cm, int ldc, int M, int N, int K, int relu,
                        hipStream_t s)
{
  const int MB = 2;
  const int Mtiles = M / 16;                 // M is always a multiple of 16
  const int groups = (Mtiles + MB - 1) / MB;
  dim3 grid((groups + 3) / 4);
  if (N == 64) {
    if (gidx) wmma_gemm_kernel<4, MB, true ><<<grid, 128, 0, s>>>(A, lda, gidx, Bp, bias, Cm, ldc, Mtiles, K, relu);
    else      wmma_gemm_kernel<4, MB, false><<<grid, 128, 0, s>>>(A, lda, gidx, Bp, bias, Cm, ldc, Mtiles, K, relu);
  } else {   // N == 112
    if (gidx) wmma_gemm_kernel<7, MB, true ><<<grid, 128, 0, s>>>(A, lda, gidx, Bp, bias, Cm, ldc, Mtiles, K, relu);
    else      wmma_gemm_kernel<7, MB, false><<<grid, 128, 0, s>>>(A, lda, gidx, Bp, bias, Cm, ldc, Mtiles, K, relu);
  }
}

// ---------------------------------------------------------------------------
// Input flattening assumption: top-level dict insertion order
//   [0]=x [1]=edge_index [2]=edge_attr [3]=y [4]=mask, then per-layer params
//   in dict insertion order: Wq,bq,Wkn,bkn,Wv,bv,Wke,bke,Wskip,bskip,
//                            emb,Wltk,bltk,Wkto,bkto,Wcomb,bcomb  (17/layer)
// ---------------------------------------------------------------------------
extern "C" void kernel_launch(void* const* d_in, const int* in_sizes, int n_in,
                              void* d_out, int out_size, void* d_ws, size_t ws_size,
                              hipStream_t stream)
{
  (void)in_sizes; (void)n_in; (void)out_size; (void)ws_size;

  const float* x    = (const float*)d_in[0];
  const int*   ei   = (const int*)  d_in[1];
  const float* ea   = (const float*)d_in[2];
  const float* y    = (const float*)d_in[3];
  const int*   mask = (const int*)  d_in[4];
  const int* src  = ei;
  const int* dstp = ei + NEDGE;

  // workspace carve (fp32)
  float* ws = (float*)d_ws;
  float* q     = ws; ws += (size_t)NNODE * LDIM;
  float* kb    = ws; ws += (size_t)NNODE * LDIM;
  float* vb    = ws; ws += (size_t)NNODE * LDIM;
  float* ekb   = ws; ws += (size_t)NEDGE * LDIM;
  float* tb    = ws; ws += (size_t)NEDGE * LDIM;
  float* sfeat = ws; ws += NEDGE;
  float* eexp  = ws; ws += NEDGE;
  float* xlb   = ws; ws += (size_t)NEDGE * LK;
  float* denb  = ws; ws += NNODE;
  unsigned* nmax = (unsigned*)ws; ws += NNODE;
  float* h2    = ws; ws += (size_t)NNODE * 3 * LDIM;
  float* ha    = ws; ws += (size_t)NNODE * LDIM;
  float* hb    = ws; ws += (size_t)NNODE * LDIM;
  _Float16* arena = (_Float16*)ws;        // packed-weight arena (bump alloc)

  const int inD[3]  = {8, 64, 64};
  const int outD[3] = {64, 64, 112};

  const float* X = x;
  for (int L = 0; L < 3; ++L) {
    const int I = inD[L], C = outD[L];
    const int pb = 5 + 17 * L;
    const float* Wq    = (const float*)d_in[pb + 0];
    const float* bq    = (const float*)d_in[pb + 1];
    const float* Wkn   = (const float*)d_in[pb + 2];
    const float* bkn   = (const float*)d_in[pb + 3];
    const float* Wv    = (const float*)d_in[pb + 4];
    const float* bv    = (const float*)d_in[pb + 5];
    const float* Wke   = (const float*)d_in[pb + 6];
    const float* bke   = (const float*)d_in[pb + 7];
    const float* Wskip = (const float*)d_in[pb + 8];
    const float* bskip = (const float*)d_in[pb + 9];
    const float* emb   = (const float*)d_in[pb + 10];
    const float* Wltk  = (const float*)d_in[pb + 11];
    const float* bltk  = (const float*)d_in[pb + 12];
    const float* Wkto  = (const float*)d_in[pb + 13];
    const float* bkto  = (const float*)d_in[pb + 14];
    const float* Wcomb = (const float*)d_in[pb + 15];
    const float* bcomb = (const float*)d_in[pb + 16];

    // pack this layer's GEMM weights (tiny; L2-resident)
    _Float16* pWq    = pack_weight(Wq,    C, 0, I,     C,    arena, stream);
    _Float16* pWkn   = pack_weight(Wkn,   C, 0, I,     C,    arena, stream);
    _Float16* pWv    = pack_weight(Wv,    C, 0, I,     C,    arena, stream);
    _Float16* pWskip = pack_weight(Wskip, C, 0, I,     C,    arena, stream);
    _Float16* pWke   = pack_weight(Wke,   C, 0, 8,     C,    arena, stream);
    _Float16* pEmbT  = pack_weight(emb,   C, 1, C,     LDIM, arena, stream); // emb^T
    _Float16* pWcomb = pack_weight(Wcomb, C, 0, 3 * C, C,    arena, stream);

    // q/k/v node GEMMs
    launch_gemm(X, I, nullptr, pWq,  bq,  q,  C, NNODE, C, I, 0, stream);
    launch_gemm(X, I, nullptr, pWkn, bkn, kb, C, NNODE, C, I, 0, stream);
    launch_gemm(X, I, nullptr, pWv,  bv,  vb, C, NNODE, C, I, 0, stream);

    // h2 = [skip | f_sum | lmsg_sum]; zero then write skip block
    hipMemsetAsync(h2, 0, (size_t)NNODE * 3 * C * sizeof(float), stream);
    launch_gemm(X, I, nullptr, pWskip, bskip, h2, 3 * C, NNODE, C, I, 0, stream);

    // ek = edge_attr @ Wke + bke
    launch_gemm(ea, 8, nullptr, pWke, bke, ekb, C, NEDGE, C, 8, 0, stream);

    // t = q[dst] @ emb^T  (row-gathered A, WMMA)
    launch_gemm(q, C, dstp, pEmbT, nullptr, tb, LDIM, NEDGE, LDIM, C, 0, stream);

    // segment-softmax state
    fill_u32_kernel<<<(NNODE + 255) / 256, 256, 0, stream>>>(nmax, 0x007FFFFFu, NNODE);
    hipMemsetAsync(denb, 0, NNODE * sizeof(float), stream);

    edge_scores_kernel<<<(NEDGE + 7) / 8, 256, 0, stream>>>(
        src, dstp, q, kb, ekb, y, tb, Wltk, bltk, mask, sfeat, xlb, nmax, NEDGE, C);
    edge_expsum_kernel<<<(NEDGE + 255) / 256, 256, 0, stream>>>(
        dstp, sfeat, nmax, eexp, denb, NEDGE);
    edge_scatter_kernel<<<(NEDGE + 7) / 8, 256, 0, stream>>>(
        src, dstp, vb, eexp, denb, xlb, Wkto, bkto, h2, 3 * C, C, NEDGE);

    // combine: out = h2 @ Wcomb + bcomb (+relu on layers 0,1)
    float* OUT = (L == 2) ? (float*)d_out : ((L == 0) ? ha : hb);
    launch_gemm(h2, 3 * C, nullptr, pWcomb, bcomb, OUT, C, NNODE, C, 3 * C,
                (L < 2) ? 1 : 0, stream);
    X = OUT;
  }
}
